// FlowMLError_16853451670048
// MI455X (gfx1250) — matrix operand order repaired and checked by reading, not compiled, not run on hardware
//
#include <hip/hip_runtime.h>
#include <stdint.h>
#include <stddef.h>

typedef __attribute__((ext_vector_type(16))) _Float16 v16h;
typedef __attribute__((ext_vector_type(8)))  float    v8f;

#define KKNN 6
#define BIGF 1e10f

__device__ __forceinline__ float lrelu(float v) { return v >= 0.f ? v : 0.1f * v; }

// Sorted-ascending top-6 insert; all indices static after unroll -> stays in registers.
__device__ __forceinline__ void topk6_insert(float d, int idx, float* bd, int* bi) {
  if (d < bd[5]) {
    bd[5] = d; bi[5] = idx;
#pragma unroll
    for (int j = 5; j > 0; --j) {
      if (bd[j] < bd[j - 1]) {
        float td = bd[j]; bd[j] = bd[j - 1]; bd[j - 1] = td;
        int   ti = bi[j]; bi[j] = bi[j - 1]; bi[j - 1] = ti;
      }
    }
  }
}

// ---------------------------------------------------------------------------
// Feature prep: f32 [N,C] -> f16 [N,Cp] zero padded, plus f32 squared norms.
// ---------------------------------------------------------------------------
__global__ void prep_half(const float* __restrict__ X, int N, int C, int Cp,
                          _Float16* __restrict__ Xh, float* __restrict__ sq) {
  int i = blockIdx.x * blockDim.x + threadIdx.x;
  if (i >= N) return;
  float s = 0.f;
  for (int c = 0; c < Cp; ++c) {
    float v = (c < C) ? X[(size_t)i * C + c] : 0.f;
    if (c < C) s += v * v;
    Xh[(size_t)i * Cp + c] = (_Float16)v;
  }
  sq[i] = s;
}

// ---------------------------------------------------------------------------
// KNN via WMMA Gram tiles. NKC = Cp/32 (1, 2, or 6).
// Block = 256 threads = 8 waves; wave w owns row tile (blockIdx*8+w)*16.
// Each staging round covers 4 column tiles (64 columns) to amortize barriers.
// ---------------------------------------------------------------------------
template <int NKC>
__global__ void __launch_bounds__(256)
knn_kernel(const _Float16* __restrict__ Xh, const float* __restrict__ sq,
           int* __restrict__ knn, int N) {
  const int Cp = NKC * 32;
  __shared__ float tile[8][16][68];      // 16 rows x 64 cols, padded stride
  __shared__ float smd[8][16][KKNN];
  __shared__ int   smi[8][16][KKNN];

  const int wv = threadIdx.x >> 5;
  const int lane = threadIdx.x & 31;
  const int rm = lane & 15;              // row (A) / col (B,C) index within tile
  const int hs = lane >> 4;              // half select
  const int rowBase = (blockIdx.x * 8 + wv) * 16;

  // --- A fragments: lane holds row (rowBase+rm); ISA 16-bit A 16x32 layout:
  // elements 0..7  = K = kc*32 + hs*8 + (0..7)
  // elements 8..15 = K = kc*32 + 16 + hs*8 + (0..7)
  int ar = rowBase + rm; if (ar > N - 1) ar = N - 1;
  const _Float16* arow = Xh + (size_t)ar * Cp;
  v16h a[NKC];
#pragma unroll
  for (int kc = 0; kc < NKC; ++kc) {
#pragma unroll
    for (int e = 0; e < 8; ++e) {
      a[kc][e]     = arow[kc * 32 + hs * 8 + e];
      a[kc][8 + e] = arow[kc * 32 + 16 + hs * 8 + e];
    }
  }
  // Row norms for the 8 rows this lane's C fragment covers (m = hs*8 + r8).
  float rsq[8];
#pragma unroll
  for (int r8 = 0; r8 < 8; ++r8) {
    int rr = rowBase + hs * 8 + r8;
    rsq[r8] = (rr < N) ? sq[rr] : 0.f;
  }

  float bd[KKNN]; int bi[KKNN];
#pragma unroll
  for (int j = 0; j < KKNN; ++j) { bd[j] = BIGF; bi[j] = 0; }

  const int ntiles = (N + 15) >> 4;
  const int ngroups = (ntiles + 3) >> 2;
  for (int g = 0; g < ngroups; ++g) {
    const int gBase = g * 64;

    // Prefetch next group's column rows (sequential stream).
    {
      int pr = gBase + 64 + lane;
      if (pr > N - 1) pr = N - 1;
      __builtin_prefetch(Xh + (size_t)pr * Cp, 0, 1);
    }

#pragma unroll
    for (int u = 0; u < 4; ++u) {
      const int colBase = gBase + u * 16;
      int cr = colBase + rm;
      int cc = (cr > N - 1) ? (N - 1) : cr;
      const _Float16* brow = Xh + (size_t)cc * Cp;
      v8f acc = {};
#pragma unroll
      for (int kc = 0; kc < NKC; ++kc) {
        // ISA 16-bit B 32x16 layout: lane holds column rm, K = kc*32 + hs*8 + e (e<8), kc*32 + 16 + hs*8 + (e-8) (e>=8).
        v16h b;
#pragma unroll
        for (int e = 0; e < 16; ++e) b[e] = brow[kc * 32 + ((e < 8) ? (hs * 8 + e) : (16 + hs * 8 + e - 8))];
        acc = __builtin_amdgcn_wmma_f32_16x16x32_f16(false, a[kc], false, b,
                                                     (short)0, acc, false, false);
      }
      float csq = sq[cc];   // clamped unconditional load; tail masked below
#pragma unroll
      for (int r8 = 0; r8 < 8; ++r8) {
        int m = hs * 8 + r8;
        int row = rowBase + m;
        float d = rsq[r8] + csq - 2.f * acc[r8];
        if (cr >= N || cr == row) d = BIGF;   // OOB column / self-loop mask
        tile[wv][m][u * 16 + rm] = d;
      }
    }
    __syncthreads();
    // Lane handles its own row rm; halves split the 64 staged columns.
#pragma unroll
    for (int q = 0; q < 32; ++q)
      topk6_insert(tile[wv][rm][hs * 32 + q], gBase + hs * 32 + q, bd, bi);
    __syncthreads();
  }

  // Merge the two half-lanes' top-6 for each row, write result.
  if (hs == 1) {
#pragma unroll
    for (int j = 0; j < KKNN; ++j) { smd[wv][rm][j] = bd[j]; smi[wv][rm][j] = bi[j]; }
  }
  __syncthreads();
  if (hs == 0) {
#pragma unroll
    for (int j = 0; j < KKNN; ++j) topk6_insert(smd[wv][rm][j], smi[wv][rm][j], bd, bi);
    int row = rowBase + rm;
    if (row < N) {
#pragma unroll
      for (int j = 0; j < KKNN; ++j) knn[row * KKNN + j] = bi[j];
    }
  }
}

// ---------------------------------------------------------------------------
// EdgeConv stage 1: H1[e] = lrelu(concat(xi, xj-xi) @ W1 + b1), per edge.
// ---------------------------------------------------------------------------
__global__ void edge_mlp1(const float* __restrict__ X, int cin,
                          const int* __restrict__ knn,
                          const float* __restrict__ W1, const float* __restrict__ b1,
                          float* __restrict__ H1, int NE) {
  __shared__ float sW[12288];   // max 2*192*32
  __shared__ float sb[32];
  const int nw = 2 * cin * 32;
  for (int q = threadIdx.x; q < nw; q += 256) sW[q] = W1[q];
  if (threadIdx.x < 32) sb[threadIdx.x] = b1[threadIdx.x];
  __syncthreads();
  int e = blockIdx.x * 256 + threadIdx.x;
  if (e >= NE) return;
  int tgt = e / KKNN;
  int src = knn[e];
  float h[32];
#pragma unroll
  for (int j = 0; j < 32; ++j) h[j] = sb[j];
  for (int c = 0; c < cin; ++c) {
    float xi = X[(size_t)tgt * cin + c];
    float dx = X[(size_t)src * cin + c] - xi;
    const float* w0 = sW + c * 32;
    const float* w1 = sW + (cin + c) * 32;
#pragma unroll
    for (int j = 0; j < 32; ++j) h[j] += xi * w0[j] + dx * w1[j];
  }
#pragma unroll
  for (int j = 0; j < 32; ++j) H1[(size_t)e * 32 + j] = lrelu(h[j]);
}

// ---------------------------------------------------------------------------
// Deterministic single-block BN reduction: stats = {mean[32], rstd[32]}.
// ---------------------------------------------------------------------------
__global__ void bn_reduce(const float* __restrict__ H, int E, float* __restrict__ stats) {
  __shared__ float red[256 * 32];
  float s[32], q[32];
#pragma unroll
  for (int j = 0; j < 32; ++j) { s[j] = 0.f; q[j] = 0.f; }
  for (int e = threadIdx.x; e < E; e += 256) {
#pragma unroll
    for (int j = 0; j < 32; ++j) {
      float v = H[(size_t)e * 32 + j];
      s[j] += v; q[j] += v * v;
    }
  }
  const float invE = 1.f / (float)E;
#pragma unroll
  for (int j = 0; j < 32; ++j) red[threadIdx.x * 32 + j] = s[j];
  __syncthreads();
  for (int st = 128; st > 0; st >>= 1) {
    if (threadIdx.x < st) {
#pragma unroll
      for (int j = 0; j < 32; ++j)
        red[threadIdx.x * 32 + j] += red[(threadIdx.x + st) * 32 + j];
    }
    __syncthreads();
  }
  float mean = 0.f;
  if (threadIdx.x < 32) { mean = red[threadIdx.x] * invE; stats[threadIdx.x] = mean; }
  __syncthreads();
#pragma unroll
  for (int j = 0; j < 32; ++j) red[threadIdx.x * 32 + j] = q[j];
  __syncthreads();
  for (int st = 128; st > 0; st >>= 1) {
    if (threadIdx.x < st) {
#pragma unroll
      for (int j = 0; j < 32; ++j)
        red[threadIdx.x * 32 + j] += red[(threadIdx.x + st) * 32 + j];
    }
    __syncthreads();
  }
  if (threadIdx.x < 32) {
    float var = red[threadIdx.x] * invE - mean * mean;
    stats[32 + threadIdx.x] = rsqrtf(var + 1e-5f);
  }
}

// ---------------------------------------------------------------------------
// EdgeConv stage 2: H2[e] = lrelu(BN(H1[e]) @ W2 + b2).
// ---------------------------------------------------------------------------
__global__ void edge_mlp2(const float* __restrict__ H1, const float* __restrict__ stats,
                          const float* __restrict__ g, const float* __restrict__ be,
                          const float* __restrict__ W2, const float* __restrict__ b2,
                          float* __restrict__ H2, int NE) {
  __shared__ float sW[1024], sA[32], sB[32], sb2[32];
  for (int q = threadIdx.x; q < 1024; q += 256) sW[q] = W2[q];
  if (threadIdx.x < 32) {
    float al = stats[32 + threadIdx.x] * g[threadIdx.x];
    sA[threadIdx.x] = al;
    sB[threadIdx.x] = be[threadIdx.x] - stats[threadIdx.x] * al;
    sb2[threadIdx.x] = b2[threadIdx.x];
  }
  __syncthreads();
  int e = blockIdx.x * 256 + threadIdx.x;
  if (e >= NE) return;
  float t[32];
#pragma unroll
  for (int c = 0; c < 32; ++c) t[c] = sA[c] * H1[(size_t)e * 32 + c] + sB[c];
#pragma unroll
  for (int j = 0; j < 32; ++j) {
    float o = sb2[j];
#pragma unroll
    for (int c = 0; c < 32; ++c) o += t[c] * sW[c * 32 + j];
    H2[(size_t)e * 32 + j] = lrelu(o);
  }
}

// ---------------------------------------------------------------------------
// EdgeConv stage 3 + aggregation:
// out[i] = sum_{j<6}(BN(H2[e])@W3 + b3) = (sum_j BN(H2))@W3 + 6*b3
// ---------------------------------------------------------------------------
__global__ void edge_agg(const float* __restrict__ H2, const float* __restrict__ stats,
                         const float* __restrict__ g, const float* __restrict__ be,
                         const float* __restrict__ W3, const float* __restrict__ b3,
                         int cout, float* __restrict__ out, int N) {
  __shared__ float sW[4096], sA[32], sB[32];   // max 32*128 weights
  const int nw = 32 * cout;
  for (int q = threadIdx.x; q < nw; q += 256) sW[q] = W3[q];
  if (threadIdx.x < 32) {
    float al = stats[32 + threadIdx.x] * g[threadIdx.x];
    sA[threadIdx.x] = al;
    sB[threadIdx.x] = be[threadIdx.x] - stats[threadIdx.x] * al;
  }
  __syncthreads();
  int i = blockIdx.x * 256 + threadIdx.x;
  if (i >= N) return;
  float ts[32];
#pragma unroll
  for (int c = 0; c < 32; ++c) ts[c] = 0.f;
  for (int j = 0; j < KKNN; ++j) {
    const float* h = H2 + (size_t)(i * KKNN + j) * 32;
#pragma unroll
    for (int c = 0; c < 32; ++c) ts[c] += sA[c] * h[c] + sB[c];
  }
  for (int o = 0; o < cout; ++o) {
    float v = 6.f * b3[o];
#pragma unroll
    for (int c = 0; c < 32; ++c) v += ts[c] * sW[c * cout + o];
    out[(size_t)i * cout + o] = v;
  }
}

// ---------------------------------------------------------------------------
// Misc glue kernels.
// ---------------------------------------------------------------------------
__global__ void concat_app(const float* __restrict__ app, const float* __restrict__ h128,
                           float* __restrict__ out, int N) {
  int t = blockIdx.x * 256 + threadIdx.x;
  if (t >= N * 192) return;
  int i = t / 192, c = t % 192;
  out[t] = (c < 64) ? app[(size_t)i * 64 + c] : h128[(size_t)i * 128 + (c - 64)];
}

__global__ void build_g4(const float* __restrict__ x, const float* __restrict__ hcol,
                         float* __restrict__ out, int N) {
  int i = blockIdx.x * 256 + threadIdx.x;
  if (i >= N) return;
  out[i * 4 + 0] = x[i * 3 + 0];
  out[i * 4 + 1] = x[i * 3 + 1];
  out[i * 4 + 2] = x[i * 3 + 2];
  out[i * 4 + 3] = hcol[i];
}

// Degree (incl. self loop) -> dinv, via LDS-staged edge scan. Deterministic.
__global__ void compute_dinv(const int* __restrict__ ei, int E,
                             float* __restrict__ dinv, int N) {
  __shared__ int sT[1024];
  int i = blockIdx.x * 256 + threadIdx.x;
  int cnt = 1;
  for (int base = 0; base < E; base += 1024) {
    int c = E - base; if (c > 1024) c = 1024;
    __syncthreads();
    for (int q = threadIdx.x; q < c; q += 256) sT[q] = ei[E + base + q];
    __syncthreads();
    for (int q = 0; q < c; ++q) cnt += (sT[q] == i) ? 1 : 0;
  }
  if (i < N) dinv[i] = rsqrtf((float)cnt);
}

// xw = X @ W, W staged in LDS, accumulators in registers.
template <int COUT>
__global__ void gcn_xw(const float* __restrict__ X, const float* __restrict__ W,
                       int cin, float* __restrict__ xw, int N) {
  __shared__ float sW[64 * COUT];
  const int nw = cin * COUT;
  for (int q = threadIdx.x; q < nw; q += 256) sW[q] = W[q];
  __syncthreads();
  int i = blockIdx.x * 256 + threadIdx.x;
  if (i >= N) return;
  float acc[COUT];
#pragma unroll
  for (int j = 0; j < COUT; ++j) acc[j] = 0.f;
  for (int c = 0; c < cin; ++c) {
    float xv = X[(size_t)i * cin + c];
#pragma unroll
    for (int j = 0; j < COUT; ++j) acc[j] += xv * sW[c * COUT + j];
  }
#pragma unroll
  for (int j = 0; j < COUT; ++j) xw[(size_t)i * COUT + j] = acc[j];
}

// out[i] = [lrelu]( dinv_i^2*xw[i] + sum_{e: tgt=i} dinv_i*dinv_src*xw[src] + b )
template <int COUT, bool RELU>
__global__ void gcn_agg(const float* __restrict__ xw, const int* __restrict__ ei, int E,
                        const float* __restrict__ dinv, const float* __restrict__ bias,
                        float* __restrict__ out, int N) {
  __shared__ int sS[1024], sT[1024];
  int i = blockIdx.x * 256 + threadIdx.x;
  bool act = i < N;
  int ii = act ? i : 0;
  float di = act ? dinv[ii] : 0.f;
  float acc[COUT];
#pragma unroll
  for (int c = 0; c < COUT; ++c) acc[c] = di * di * xw[(size_t)ii * COUT + c];
  for (int base = 0; base < E; base += 1024) {
    int cnt = E - base; if (cnt > 1024) cnt = 1024;
    __syncthreads();
    for (int q = threadIdx.x; q < cnt; q += 256) {
      sS[q] = ei[base + q];
      sT[q] = ei[E + base + q];
    }
    __syncthreads();
    for (int q = 0; q < cnt; ++q) {
      if (sT[q] == i) {
        int s = sS[q];
        float nrm = di * dinv[s];
#pragma unroll
        for (int c = 0; c < COUT; ++c) acc[c] += nrm * xw[(size_t)s * COUT + c];
      }
    }
  }
  if (act) {
#pragma unroll
    for (int c = 0; c < COUT; ++c) {
      float o = acc[c] + bias[c];
      if (RELU) o = lrelu(o);
      out[(size_t)i * COUT + c] = o;
    }
  }
}

// ---------------------------------------------------------------------------
// Host-side orchestration.
// ---------------------------------------------------------------------------
struct ECp { const float *W1, *b1, *g1, *be1, *W2, *b2, *g2, *be2, *W3, *b3; };
struct GCp { const float *W, *b; };

static void run_knn(const float* X, int C, int N, _Float16* Xh, float* sq, int* knn,
                    hipStream_t s) {
  int Cp = (C + 31) & ~31;
  prep_half<<<(N + 255) / 256, 256, 0, s>>>(X, N, C, Cp, Xh, sq);
  int rowTiles = (N + 15) / 16;
  int blocks = (rowTiles + 7) / 8;
  switch (Cp / 32) {
    case 1: knn_kernel<1><<<blocks, 256, 0, s>>>(Xh, sq, knn, N); break;
    case 2: knn_kernel<2><<<blocks, 256, 0, s>>>(Xh, sq, knn, N); break;
    default: knn_kernel<6><<<blocks, 256, 0, s>>>(Xh, sq, knn, N); break;
  }
}

static void run_edgeconv(const float* X, int cin, int cout, int N, const int* knn,
                         const ECp& p, float* H1, float* H2, float* st1, float* st2,
                         float* out, hipStream_t s) {
  int NE = N * KKNN;
  int eb = (NE + 255) / 256;
  edge_mlp1<<<eb, 256, 0, s>>>(X, cin, knn, p.W1, p.b1, H1, NE);
  bn_reduce<<<1, 256, 0, s>>>(H1, NE, st1);
  edge_mlp2<<<eb, 256, 0, s>>>(H1, st1, p.g1, p.be1, p.W2, p.b2, H2, NE);
  bn_reduce<<<1, 256, 0, s>>>(H2, NE, st2);
  edge_agg<<<(N + 255) / 256, 256, 0, s>>>(H2, st2, p.g2, p.be2, p.W3, p.b3, cout, out, N);
}

extern "C" void kernel_launch(void* const* d_in, const int* in_sizes, int n_in,
                              void* d_out, int out_size, void* d_ws, size_t ws_size,
                              hipStream_t stream) {
  (void)n_in; (void)out_size; (void)ws_size;
  const float* x   = (const float*)d_in[0];
  const float* pos = (const float*)d_in[1];
  const int*   ei  = (const int*)d_in[2];
  const int N = in_sizes[1] / 2;
  const int E = in_sizes[2] / 2;

  ECp ec[5]; GCp gc[5];
  int pi = 4;
  for (int l = 0; l < 5; ++l) {
    const float* f[10];
    for (int k = 0; k < 10; ++k) f[k] = (const float*)d_in[pi++];
    ec[l] = ECp{f[0], f[1], f[2], f[3], f[4], f[5], f[6], f[7], f[8], f[9]};
  }
  for (int l = 0; l < 5; ++l) {
    gc[l].W = (const float*)d_in[pi++];
    gc[l].b = (const float*)d_in[pi++];
  }

  uint8_t* w = (uint8_t*)d_ws;
  auto take = [&](size_t bytes) -> uint8_t* {
    uint8_t* r = w; w += (bytes + 255) & ~(size_t)255; return r;
  };
  _Float16* bXh = (_Float16*)take((size_t)N * 192 * 2);
  float* bSq    = (float*)take((size_t)N * 4);
  int*   bKnn   = (int*)take((size_t)N * KKNN * 4);
  float* bApp   = (float*)take((size_t)N * 64 * 4);
  float* bF1    = (float*)take((size_t)N * 192 * 4);
  float* bF2    = (float*)take((size_t)N * 192 * 4);
  float* bH1    = (float*)take((size_t)N * KKNN * 32 * 4);
  float* bH2    = (float*)take((size_t)N * KKNN * 32 * 4);
  float* bSt1   = (float*)take(64 * 4);
  float* bSt2   = (float*)take(64 * 4);
  float* bDinv  = (float*)take((size_t)N * 4);
  float* bXw    = (float*)take((size_t)N * 64 * 4);

  const int nb = (N + 255) / 256;

  // ec1: pos(2) -> append(64)
  run_knn(pos, 2, N, bXh, bSq, bKnn, stream);
  run_edgeconv(pos, 2, 64, N, bKnn, ec[0], bH1, bH2, bSt1, bSt2, bApp, stream);
  // eca: append -> bF1(64)
  run_knn(bApp, 64, N, bXh, bSq, bKnn, stream);
  run_edgeconv(bApp, 64, 64, N, bKnn, ec[1], bH1, bH2, bSt1, bSt2, bF1, stream);
  // ecb: bF1 -> bF2(64)
  run_knn(bF1, 64, N, bXh, bSq, bKnn, stream);
  run_edgeconv(bF1, 64, 64, N, bKnn, ec[2], bH1, bH2, bSt1, bSt2, bF2, stream);
  // ec2: bF2 -> bF1(128)
  run_knn(bF2, 64, N, bXh, bSq, bKnn, stream);
  run_edgeconv(bF2, 64, 128, N, bKnn, ec[3], bH1, bH2, bSt1, bSt2, bF1, stream);
  // ec3: concat(append, 128) = bF2(192) -> bSq(1)  (bSq free after last knn)
  concat_app<<<(N * 192 + 255) / 256, 256, 0, stream>>>(bApp, bF1, bF2, N);
  run_knn(bF2, 192, N, bXh, bSq, bKnn, stream);
  run_edgeconv(bF2, 192, 1, N, bKnn, ec[4], bH1, bH2, bSt1, bSt2, bSq, stream);

  // GCN stack on fixed edge_index.
  build_g4<<<nb, 256, 0, stream>>>(x, bSq, bF1, N);
  compute_dinv<<<nb, 256, 0, stream>>>(ei, E, bDinv, N);

  gcn_xw<64><<<nb, 256, 0, stream>>>(bF1, gc[0].W, 4, bXw, N);
  gcn_agg<64, true><<<nb, 256, 0, stream>>>(bXw, ei, E, bDinv, gc[0].b, bF2, N);
  gcn_xw<64><<<nb, 256, 0, stream>>>(bF2, gc[1].W, 64, bXw, N);
  gcn_agg<64, true><<<nb, 256, 0, stream>>>(bXw, ei, E, bDinv, gc[1].b, bF1, N);
  gcn_xw<64><<<nb, 256, 0, stream>>>(bF1, gc[2].W, 64, bXw, N);
  gcn_agg<64, true><<<nb, 256, 0, stream>>>(bXw, ei, E, bDinv, gc[2].b, bF2, N);
  gcn_xw<64><<<nb, 256, 0, stream>>>(bF2, gc[3].W, 64, bXw, N);
  gcn_agg<64, true><<<nb, 256, 0, stream>>>(bXw, ei, E, bDinv, gc[3].b, bF1, N);
  gcn_xw<3><<<nb, 256, 0, stream>>>(bF1, gc[4].W, 64, bXw, N);
  gcn_agg<3, false><<<nb, 256, 0, stream>>>(bXw, ei, E, bDinv, gc[4].b, (float*)d_out, N);
}
